// RNNDecoderHat_70351564309221
// MI455X (gfx1250) — matrix operand order repaired and checked by reading, not compiled, not run on hardware
//
#include <hip/hip_runtime.h>
#include <hip/hip_bf16.h>

typedef __attribute__((ext_vector_type(16))) __bf16 v16bf;
typedef __attribute__((ext_vector_type(8)))  float  v8f;

union Frag32B { uint4 q[2]; v16bf v; };

#define WMMA_BF16(A, B, C) \
  __builtin_amdgcn_wmma_f32_16x16x32_bf16(false, (A), false, (B), (short)0, (C), false, false)

// ---- constants ----
#define BATCH 4096
#define HDIM  256
#define TLEN  128
#define VOUT  4
#define WSTR  264            // bf16 row stride: 256 + 8 pad -> 528B, 528 % 256 == 16 (bank-friendly)
#define ROWS_PER_WAVE 16
#define WAVES_PER_BLK 2
#define THREADS (WAVES_PER_BLK * 32)
#define NBLOCKS (BATCH / (WAVES_PER_BLK * ROWS_PER_WAVE))
// LDS layout (bf16 elements): W[256*WSTR] | Wout[16*WSTR] | H[WAVES*16*WSTR]
#define LDS_W_ELEMS   (256 * WSTR)
#define LDS_O_ELEMS   (16 * WSTR)
#define LDS_H_ELEMS   (ROWS_PER_WAVE * WSTR)
#define LDS_BYTES     ((LDS_W_ELEMS + LDS_O_ELEMS + WAVES_PER_BLK * LDS_H_ELEMS) * 2)

__device__ __forceinline__ unsigned bf16pk2(float lo, float hi) {
  unsigned ul = __float_as_uint(lo), uh = __float_as_uint(hi);
  ul = (ul + 0x7FFFu + ((ul >> 16) & 1u)) >> 16;
  uh = (uh + 0x7FFFu + ((uh >> 16) & 1u)) >> 16;
  return ul | (uh << 16);
}
__device__ __forceinline__ unsigned short bf16pk1(float x) {
  unsigned u = __float_as_uint(x);
  return (unsigned short)((u + 0x7FFFu + ((u >> 16) & 1u)) >> 16);
}

// hardware tanh if gfx1250 exposes it; otherwise exp2/rcp formula (saturates, no NaN)
__device__ __forceinline__ float fast_tanh(float x) {
#if __has_builtin(__builtin_amdgcn_tanhf)
  return __builtin_amdgcn_tanhf(x);
#else
  const float e = __builtin_amdgcn_exp2f(x * 2.8853900817779268f);  // 2*log2(e)
  return 1.0f - 2.0f * __builtin_amdgcn_rcpf(e + 1.0f);
#endif
}

// intra-quad lane exchanges on the VALU pipe (DPP quad_perm), no LDS traffic
__device__ __forceinline__ float quad_xor1(float x) {
#if __has_builtin(__builtin_amdgcn_mov_dpp)
  return __int_as_float(__builtin_amdgcn_mov_dpp(__float_as_int(x), 0xB1, 0xF, 0xF, true));
#else
  return __shfl_xor(x, 1, 32);
#endif
}
__device__ __forceinline__ float quad_xor2(float x) {
#if __has_builtin(__builtin_amdgcn_mov_dpp)
  return __int_as_float(__builtin_amdgcn_mov_dpp(__float_as_int(x), 0x4E, 0xF, 0xF, true));
#else
  return __shfl_xor(x, 2, 32);
#endif
}

// A fragment (16x32 bf16): lane m(0-15): K = kt*32+[0..7],[16..23]; lane m+16: +8
__device__ __forceinline__ v16bf load_afrag(const unsigned short* hbuf, int lane, int kt) {
  const int m = lane & 15, sel = (lane >> 4) & 1;
  const unsigned short* p = hbuf + m * WSTR + kt * 32 + sel * 8;
  Frag32B f;
  f.q[0] = *(const uint4*)(p);
  f.q[1] = *(const uint4*)(p + 16);
  return f.v;
}

// B fragment (32x16 bf16): lane n(0-15): K = kt*32+[0..7],[16..23]; lane n+16: +8
// B[k][n] = W[n][k]  (W row-major bf16, stride WSTR) -> contiguous per lane.
__device__ __forceinline__ v16bf load_bfrag(const unsigned short* wbuf, int lane, int kt, int nt) {
  const int n = nt * 16 + (lane & 15), sel = (lane >> 4) & 1;
  const unsigned short* p = wbuf + n * WSTR + kt * 32 + sel * 8;
  Frag32B f;
  f.q[0] = *(const uint4*)(p);
  f.q[1] = *(const uint4*)(p + 16);
  return f.v;
}

// cooperative: 256x256 f32 -> bf16 LDS (stride WSTR)
__device__ __forceinline__ void stage_w(const float* __restrict__ src, unsigned short* dst,
                                        int tid, int nth) {
  for (int i = tid; i < 256 * 64; i += nth) {
    const int r = i >> 6, c4 = (i & 63) << 2;
    const float4 f = *(const float4*)(src + r * 256 + c4);
    uint2 p;
    p.x = bf16pk2(f.x, f.y);
    p.y = bf16pk2(f.z, f.w);
    *(uint2*)(dst + r * WSTR + c4) = p;
  }
}

__global__ __launch_bounds__(THREADS, 1)
void rnn_decoder_wmma(const float* __restrict__ a,
                      const float* __restrict__ W_ih,
                      const float* __restrict__ b_ih,
                      const float* __restrict__ W_hh,
                      const float* __restrict__ b_hh,
                      const float* __restrict__ W_out,
                      float* __restrict__ out) {
  extern __shared__ __align__(16) unsigned short smem[];
  unsigned short* Wbuf = smem;
  unsigned short* Obuf = Wbuf + LDS_W_ELEMS;
  unsigned short* Hall = Obuf + LDS_O_ELEMS;

  const int tid  = threadIdx.x;
  const int lane = tid & 31;
  const int wv   = tid >> 5;
  const int nth  = THREADS;
  unsigned short* hbuf = Hall + wv * LDS_H_ELEMS;
  const int rowbase = blockIdx.x * (WAVES_PER_BLK * ROWS_PER_WAVE) + wv * ROWS_PER_WAVE;

  // ---- stage this wave's 16 rows of `a` (bf16) into the h-buffer ----
  for (int i = lane; i < ROWS_PER_WAVE * 64; i += 32) {
    const int r = i >> 6, c4 = (i & 63) << 2;
    const float4 f = *(const float4*)(a + (size_t)(rowbase + r) * HDIM + c4);
    uint2 p;
    p.x = bf16pk2(f.x, f.y);
    p.y = bf16pk2(f.z, f.w);
    *(uint2*)(hbuf + r * WSTR + c4) = p;
  }
  // ---- stage W_ih into the shared weight buffer ----
  stage_w(W_ih, Wbuf, tid, nth);
  __syncthreads();

  // ---- x_proj = a @ W_ih^T + (b_ih + b_hh), kept packed bf16x2 in 64 VGPRs ----
  v16bf A[8];
#pragma unroll
  for (int kt = 0; kt < 8; ++kt) A[kt] = load_afrag(hbuf, lane, kt);

  unsigned xp_pk[16][4];
#pragma unroll
  for (int nt = 0; nt < 16; ++nt) {
    const int c = nt * 16 + (lane & 15);
    const float bias = b_ih[c] + b_hh[c];
    v8f acc;
#pragma unroll
    for (int i = 0; i < 8; ++i) acc[i] = bias;
#pragma unroll
    for (int kt = 0; kt < 8; ++kt)
      acc = WMMA_BF16(A[kt], load_bfrag(Wbuf, lane, kt, nt), acc);
#pragma unroll
    for (int j = 0; j < 4; ++j)
      xp_pk[nt][j] = bf16pk2(acc[2 * j], acc[2 * j + 1]);
  }
  __syncthreads();

  // ---- overwrite weight buffer with W_hh; stage zero-padded W_out^T ----
  stage_w(W_hh, Wbuf, tid, nth);
  for (int i = tid; i < 16 * 64; i += nth) {
    const int r = i >> 6, c4 = (i & 63) << 2;
    uint2 p;
    if (r < VOUT) {
      const float4 f = *(const float4*)(W_out + r * 256 + c4);
      p.x = bf16pk2(f.x, f.y);
      p.y = bf16pk2(f.z, f.w);
    } else {
      p.x = 0u; p.y = 0u;
    }
    *(uint2*)(Obuf + r * WSTR + c4) = p;
  }
  __syncthreads();

  // h0 = 0 -> zero A fragments (h-buffer is written before its first read)
#pragma unroll
  for (int kt = 0; kt < 8; ++kt) {
    Frag32B z;
    z.q[0] = make_uint4(0u, 0u, 0u, 0u);
    z.q[1] = make_uint4(0u, 0u, 0u, 0u);
    A[kt] = z.v;
  }

  const float L2E = 1.4426950408889634f;
  const float LN2 = 0.6931471805599453f;
  const int   nsub = lane & 15;
  const int   mhi  = (lane >> 4) << 3;  // +8 rows for upper half-wave
  const bool  storelane = nsub < VOUT;

#pragma unroll 1
  for (int t = 0; t < TLEN; ++t) {
    // ---- h_new = tanh(xp + h @ W_hh^T) : 16 N-tiles x 8 K-steps of WMMA ----
#pragma unroll
    for (int nt = 0; nt < 16; ++nt) {
      // unpack the packed-bf16 x_proj tile into the f32 accumulator
      v8f acc;
#pragma unroll
      for (int j = 0; j < 4; ++j) {
        const unsigned u = xp_pk[nt][j];
        acc[2 * j]     = __uint_as_float(u << 16);
        acc[2 * j + 1] = __uint_as_float(u & 0xFFFF0000u);
      }
      // double-buffered B-fragment stream through the K chain
      v16bf bcur = load_bfrag(Wbuf, lane, 0, nt);
#pragma unroll
      for (int kt = 0; kt < 7; ++kt) {
        const v16bf bnext = load_bfrag(Wbuf, lane, kt + 1, nt);
        acc = WMMA_BF16(A[kt], bcur, acc);
        bcur = bnext;
      }
      acc = WMMA_BF16(A[7], bcur, acc);
#pragma unroll
      for (int i = 0; i < 8; ++i) {
        const float th = fast_tanh(acc[i]);
        hbuf[(i + mhi) * WSTR + nt * 16 + nsub] = bf16pk1(th);
      }
    }
    // reload A fragments with h_new (DS ops are in-order within a wave)
#pragma unroll
    for (int kt = 0; kt < 8; ++kt) A[kt] = load_afrag(hbuf, lane, kt);

    // ---- y = h_new @ W_out^T  (one padded N-tile, double-buffered) ----
    v8f y = {};
    {
      v16bf bcur = load_bfrag(Obuf, lane, 0, 0);
#pragma unroll
      for (int kt = 0; kt < 7; ++kt) {
        const v16bf bnext = load_bfrag(Obuf, lane, kt + 1, 0);
        y = WMMA_BF16(A[kt], bcur, y);
        bcur = bnext;
      }
      y = WMMA_BF16(A[7], bcur, y);
    }

    // ---- log_softmax over V=4 (DPP quad butterfly, VALU-only) + store ----
#pragma unroll
    for (int i = 0; i < 8; ++i) {
      const float v  = y[i];
      float m1 = fmaxf(v, quad_xor1(v));
      float mx = fmaxf(m1, quad_xor2(m1));
      const float e  = __builtin_amdgcn_exp2f((v - mx) * L2E);
      float s1 = e + quad_xor1(e);
      float s  = s1 + quad_xor2(s1);
      const float lp = (v - mx) - __builtin_amdgcn_logf(s) * LN2;
      if (storelane) {
        out[((size_t)(rowbase + i + mhi) * TLEN + t) * VOUT + nsub] = lp;
      }
    }
  }
}

extern "C" void kernel_launch(void* const* d_in, const int* in_sizes, int n_in,
                              void* d_out, int out_size, void* d_ws, size_t ws_size,
                              hipStream_t stream) {
  (void)in_sizes; (void)n_in; (void)out_size; (void)d_ws; (void)ws_size;
  const float* a     = (const float*)d_in[0];
  const float* W_ih  = (const float*)d_in[1];
  const float* b_ih  = (const float*)d_in[2];
  const float* W_hh  = (const float*)d_in[3];
  const float* b_hh  = (const float*)d_in[4];
  const float* W_out = (const float*)d_in[5];
  float* out = (float*)d_out;

  // allow >64KB dynamic LDS (no-op if already permitted)
  (void)hipFuncSetAttribute((const void*)rnn_decoder_wmma,
                            hipFuncAttributeMaxDynamicSharedMemorySize, LDS_BYTES);

  rnn_decoder_wmma<<<dim3(NBLOCKS), dim3(THREADS), LDS_BYTES, stream>>>(
      a, W_ih, b_ih, W_hh, b_hh, W_out, out);
}
